// ConvGRUAutoencoder_53541062312023
// MI455X (gfx1250) — compile-verified, matrix operand order checked by reading, not run on hardware
//
#include <hip/hip_runtime.h>
#include <math.h>

typedef __attribute__((ext_vector_type(16))) _Float16 v16h;
typedef __attribute__((ext_vector_type(8)))  _Float16 v8h;
typedef __attribute__((ext_vector_type(8)))  float    v8f;

namespace {
constexpr int Bb = 64, Tt = 100, Ff = 128;
constexpr int Nn = Bb * Ff;     // 8192 positions
constexpr int NT = Nn / 16;     // 512 n-tiles
}

// ---------------- input conversion: x (B,T,F) f32 -> seqx [T][N] f16 -------
__global__ void __launch_bounds__(256)
k_convert_x(const float* __restrict__ x, _Float16* __restrict__ seqx)
{
    int idx = blockIdx.x * 256 + threadIdx.x;
    if (idx >= Bb * Tt * Ff) return;
    int f = idx % Ff;
    int t = (idx / Ff) % Tt;
    int b = idx / (Ff * Tt);
    seqx[(size_t)t * Nn + b * Ff + f] = (_Float16)x[idx];
}

__global__ void __launch_bounds__(256)
k_fill0(float* __restrict__ p, int n)
{
    int idx = blockIdx.x * 256 + threadIdx.x;
    if (idx < n) p[idx] = 0.0f;
}

// ---------------- weight packing into WMMA A-fragment layout ---------------
// frag order: (tap*nch + chunk)*Mtiles + mtile ; per-frag: [lane][j] (512 f16)
// A 16x32 f16 layout: lane L -> m = L%16, half = L/16,
//   element j -> K = (j<8 ? half*8+j : j+8+half*8)
__global__ void __launch_bounds__(256)
k_pack_w(const float* __restrict__ w, _Float16* __restrict__ dst,
         int Mtiles, int nch, int Ctot)
{
    int idx = blockIdx.x * 256 + threadIdx.x;
    int total = 3 * nch * Mtiles * 512;
    if (idx >= total) return;
    int frag = idx >> 9;
    int r    = idx & 511;
    int lane = r >> 4, j = r & 15;
    int tap = frag / (nch * Mtiles);
    int rem = frag % (nch * Mtiles);
    int ch  = rem / Mtiles, mt = rem % Mtiles;
    int half = lane >> 4;
    int m = mt * 16 + (lane & 15);
    int k = (j < 8) ? (half * 8 + j) : (j + 8 + half * 8);
    int c = ch * 32 + k;
    _Float16 v = (_Float16)0.f;
    if (c < Ctot)  // w shape (M, Ctot, 3, 3); only kw==1 contributes (width==1)
        v = (_Float16)w[((size_t)(m * Ctot + c) * 3 + tap) * 3 + 1];
    dst[idx] = v;
}

// ---------------- per-layer cat init: h-region from hstate, rest zero ------
// covers rows [0, Nn] inclusive -> row Nn is the zero guard row for boundary taps
__global__ void __launch_bounds__(256)
k_init_cat(const float* __restrict__ hstate,
           _Float16* __restrict__ cat, _Float16* __restrict__ cat2,
           int cin, int hd, int Cpad)
{
    int idx = blockIdx.x * 256 + threadIdx.x;
    if (idx >= (Nn + 1) * Cpad) return;
    int n = idx / Cpad, c = idx % Cpad;
    _Float16 v = (_Float16)0.f;
    if (n < Nn && c >= cin && c < cin + hd)
        v = (_Float16)hstate[(size_t)n * hd + (c - cin)];
    cat[idx]  = v;
    cat2[idx] = v;
}

// ---------------- per-step x copy into both cat buffers --------------------
__global__ void __launch_bounds__(256)
k_assemble_x(const _Float16* __restrict__ seqin,
             _Float16* __restrict__ cat, _Float16* __restrict__ cat2,
             int cin, int Cpad)
{
    int idx = blockIdx.x * 256 + threadIdx.x;
    if (idx >= Nn * cin) return;
    int n = idx / cin, c = idx % cin;
    _Float16 v = seqin[idx];               // seqin is [n][cin] contiguous
    size_t o = (size_t)n * Cpad + c;
    cat[o]  = v;
    cat2[o] = v;
}

// ---------------- gates GEMM: G = sigmoid(Wg * cat3) -----------------------
// two M-tiles per wave (B fragment shared); writes r*h into cat2, u into ubuf
template<int NCH>
__global__ void __launch_bounds__(256)
k_gates(const _Float16* __restrict__ cat, const _Float16* __restrict__ pw,
        const float* __restrict__ bg, const float* __restrict__ hstate,
        _Float16* __restrict__ cat2, float* __restrict__ ubuf,
        int Mtiles2, int cin, int hd)
{
    constexpr int Cpad = NCH * 32;
    const int Mtiles = Mtiles2 * 2;
    int lane  = threadIdx.x & 31;
    int wave  = threadIdx.x >> 5;
    int tile  = blockIdx.x * 8 + wave;
    int mt2   = tile % Mtiles2;
    int ntile = tile / Mtiles2;
    int half  = lane >> 4;
    int col   = lane & 15;
    int n     = ntile * 16 + col;
    int f     = n & (Ff - 1);
    int m0    = mt2 * 32;                   // first channel of the pair

    v8f acc0, acc1;
#pragma unroll
    for (int i = 0; i < 8; ++i) {
        acc0[i] = bg[m0 + 8 * half + i];
        acc1[i] = bg[m0 + 16 + 8 * half + i];
    }

#pragma unroll
    for (int tap = 0; tap < 3; ++tap) {
        const int d = tap - 1;
        const bool valid = ((unsigned)(f + d)) < (unsigned)Ff;
        const int np = valid ? (n + d) : Nn;          // row Nn = zero guard row
        const _Float16* row = cat + (size_t)np * Cpad + half * 8;
#pragma unroll
        for (int ch = 0; ch < NCH; ++ch) {
            const _Float16* bp = row + ch * 32;
            v8h blo = *(const v8h*)bp;
            v8h bhi = *(const v8h*)(bp + 16);
            const _Float16* ap = pw + (((size_t)(tap * NCH + ch) * Mtiles + mt2 * 2) << 9) + lane * 16;
            v8h a0lo = *(const v8h*)ap;
            v8h a0hi = *(const v8h*)(ap + 8);
            v8h a1lo = *(const v8h*)(ap + 512);
            v8h a1hi = *(const v8h*)(ap + 520);
            v16h a0, a1, b;
#pragma unroll
            for (int i = 0; i < 8; ++i) {
                b[i] = blo[i];    b[i + 8] = bhi[i];
                a0[i] = a0lo[i];  a0[i + 8] = a0hi[i];
                a1[i] = a1lo[i];  a1[i + 8] = a1hi[i];
            }
            acc0 = __builtin_amdgcn_wmma_f32_16x16x32_f16(false, a0, false, b,
                                                          (short)0, acc0, false, false);
            acc1 = __builtin_amdgcn_wmma_f32_16x16x32_f16(false, a1, false, b,
                                                          (short)0, acc1, false, false);
        }
    }

    // hd is a multiple of 32 -> the 32-channel pair is entirely r or entirely u
    if (m0 < hd) {   // r gates -> write r*h (f16) into cat2 h-region
#pragma unroll
        for (int i = 0; i < 8; ++i) {
            int g0 = m0 + 8 * half + i;
            int g1 = g0 + 16;
            float r0 = 1.0f / (1.0f + __expf(-acc0[i]));
            float r1 = 1.0f / (1.0f + __expf(-acc1[i]));
            float h0 = hstate[(size_t)n * hd + g0];
            float h1 = hstate[(size_t)n * hd + g1];
            cat2[(size_t)n * Cpad + cin + g0] = (_Float16)(r0 * h0);
            cat2[(size_t)n * Cpad + cin + g1] = (_Float16)(r1 * h1);
        }
    } else {         // u gates
#pragma unroll
        for (int i = 0; i < 8; ++i) {
            int c0 = m0 - hd + 8 * half + i;
            int c1 = c0 + 16;
            ubuf[(size_t)n * hd + c0] = 1.0f / (1.0f + __expf(-acc0[i]));
            ubuf[(size_t)n * hd + c1] = 1.0f / (1.0f + __expf(-acc1[i]));
        }
    }
}

// ---------------- candidate GEMM + GRU state update ------------------------
// writes h_next to hstate (f32), seqout (f16) and cat h-region (for next step)
template<int NCH>
__global__ void __launch_bounds__(256)
k_cand(const _Float16* __restrict__ cat2, const _Float16* __restrict__ pw,
       const float* __restrict__ bc, const float* __restrict__ ubuf,
       float* __restrict__ hstate, _Float16* __restrict__ seqout,
       _Float16* __restrict__ cat,
       int Mtiles2, int cin, int hd)
{
    constexpr int Cpad = NCH * 32;
    const int Mtiles = Mtiles2 * 2;
    int lane  = threadIdx.x & 31;
    int wave  = threadIdx.x >> 5;
    int tile  = blockIdx.x * 8 + wave;
    int mt2   = tile % Mtiles2;
    int ntile = tile / Mtiles2;
    int half  = lane >> 4;
    int col   = lane & 15;
    int n     = ntile * 16 + col;
    int f     = n & (Ff - 1);
    int m0    = mt2 * 32;

    v8f acc0, acc1;
#pragma unroll
    for (int i = 0; i < 8; ++i) {
        acc0[i] = bc[m0 + 8 * half + i];
        acc1[i] = bc[m0 + 16 + 8 * half + i];
    }

#pragma unroll
    for (int tap = 0; tap < 3; ++tap) {
        const int d = tap - 1;
        const bool valid = ((unsigned)(f + d)) < (unsigned)Ff;
        const int np = valid ? (n + d) : Nn;
        const _Float16* row = cat2 + (size_t)np * Cpad + half * 8;
#pragma unroll
        for (int ch = 0; ch < NCH; ++ch) {
            const _Float16* bp = row + ch * 32;
            v8h blo = *(const v8h*)bp;
            v8h bhi = *(const v8h*)(bp + 16);
            const _Float16* ap = pw + (((size_t)(tap * NCH + ch) * Mtiles + mt2 * 2) << 9) + lane * 16;
            v8h a0lo = *(const v8h*)ap;
            v8h a0hi = *(const v8h*)(ap + 8);
            v8h a1lo = *(const v8h*)(ap + 512);
            v8h a1hi = *(const v8h*)(ap + 520);
            v16h a0, a1, b;
#pragma unroll
            for (int i = 0; i < 8; ++i) {
                b[i] = blo[i];    b[i + 8] = bhi[i];
                a0[i] = a0lo[i];  a0[i + 8] = a0hi[i];
                a1[i] = a1lo[i];  a1[i + 8] = a1hi[i];
            }
            acc0 = __builtin_amdgcn_wmma_f32_16x16x32_f16(false, a0, false, b,
                                                          (short)0, acc0, false, false);
            acc1 = __builtin_amdgcn_wmma_f32_16x16x32_f16(false, a1, false, b,
                                                          (short)0, acc1, false, false);
        }
    }

#pragma unroll
    for (int i = 0; i < 8; ++i) {
        int mA = m0 + 8 * half + i;
        int mBc = mA + 16;
        float cA = tanhf(acc0[i]);
        float cB = tanhf(acc1[i]);
        size_t oA = (size_t)n * hd + mA;
        size_t oB = (size_t)n * hd + mBc;
        float uA = ubuf[oA], hA = hstate[oA];
        float uB = ubuf[oB], hB = hstate[oB];
        float hnA = (1.0f - uA) * hA + uA * cA;
        float hnB = (1.0f - uB) * hB + uB * cB;
        hstate[oA] = hnA;
        hstate[oB] = hnB;
        seqout[oA] = (_Float16)hnA;
        seqout[oB] = (_Float16)hnB;
        cat[(size_t)n * Cpad + cin + mA]  = (_Float16)hnA;   // h for next gates
        cat[(size_t)n * Cpad + cin + mBc] = (_Float16)hnB;
    }
}

// ---------------- final 1x1 conv projection --------------------------------
__global__ void __launch_bounds__(256)
k_final(const _Float16* __restrict__ z, const float* __restrict__ fw,
        const float* __restrict__ fb, float* __restrict__ out)
{
    int idx = blockIdx.x * 256 + threadIdx.x;
    if (idx >= Tt * Nn) return;
    int t = idx / Nn, n = idx % Nn;
    int b = n >> 7, f = n & 127;
    const _Float16* zp = z + (size_t)idx * 32;
    float s = fb[0];
#pragma unroll
    for (int c = 0; c < 32; ++c) s += fw[c] * (float)zp[c];
    out[((size_t)b * Tt + t) * Ff + f] = s;
}

// ===========================================================================
extern "C" void kernel_launch(void* const* d_in, const int* in_sizes, int n_in,
                              void* d_out, int out_size, void* d_ws, size_t ws_size,
                              hipStream_t stream)
{
    (void)in_sizes; (void)n_in; (void)out_size;
    const float* x = (const float*)d_in[0];
    const float* W[16];
    for (int i = 0; i < 16; ++i) W[i] = (const float*)d_in[1 + i];
    const float* fw = (const float*)d_in[17];
    const float* fb = (const float*)d_in[18];
    float* out = (float*)d_out;

    char* ws = (char*)d_ws;
    size_t off = 0;
    auto alloc = [&](size_t bytes) -> void* {
        void* p = ws + off;
        off = (off + bytes + 255) & ~(size_t)255;
        return p;
    };
    _Float16* seqx = (_Float16*)alloc((size_t)Tt * Nn * 2);        // e0 input
    _Float16* seq0 = (_Float16*)alloc((size_t)Tt * Nn * 32 * 2);   // e0 out
    _Float16* seq1 = (_Float16*)alloc((size_t)Tt * Nn * 64 * 2);   // e1 out
    _Float16* seq2 = (_Float16*)alloc((size_t)Tt * Nn * 64 * 2);   // d0 out
    _Float16* seq3 = (_Float16*)alloc((size_t)Tt * Nn * 32 * 2);   // d1 out
    _Float16* cat  = (_Float16*)alloc((size_t)(Nn + 16) * 128 * 2);
    _Float16* cat2 = (_Float16*)alloc((size_t)(Nn + 16) * 128 * 2);
    float* hstate  = (float*)alloc((size_t)Nn * 64 * 4);
    float* ubuf    = (float*)alloc((size_t)Nn * 64 * 4);
    float* h0save  = (float*)alloc((size_t)Nn * 32 * 4);
    float* h1save  = (float*)alloc((size_t)Nn * 64 * 4);
    _Float16* packs[8];
    for (int i = 0; i < 8; ++i) packs[i] = (_Float16*)alloc((size_t)3 * 4 * 8 * 1024);
    if (off > ws_size) return;  // workspace too small; cannot run

    struct Layer {
        const float *wg, *bg, *wc, *bc;
        int cin, hd, nch;
        const _Float16* sin;
        _Float16* sout;
    };
    Layer L[4] = {
        { W[0],  W[1],  W[2],  W[3],   1, 32, 2, seqx, seq0 },   // e0
        { W[4],  W[5],  W[6],  W[7],  32, 64, 3, seq0, seq1 },   // e1
        { W[8],  W[9],  W[10], W[11], 64, 64, 4, seq1, seq2 },   // d0
        { W[12], W[13], W[14], W[15], 64, 32, 3, seq2, seq3 },   // d1
    };

    k_convert_x<<<(Bb * Tt * Ff + 255) / 256, 256, 0, stream>>>(x, seqx);

    for (int l = 0; l < 4; ++l) {
        int Ctot = L[l].cin + L[l].hd;
        int MtG = 2 * L[l].hd / 16, MtC = L[l].hd / 16;
        int tg = 3 * L[l].nch * MtG * 512;
        int tc = 3 * L[l].nch * MtC * 512;
        k_pack_w<<<(tg + 255) / 256, 256, 0, stream>>>(L[l].wg, packs[2 * l],     MtG, L[l].nch, Ctot);
        k_pack_w<<<(tc + 255) / 256, 256, 0, stream>>>(L[l].wc, packs[2 * l + 1], MtC, L[l].nch, Ctot);
    }

    for (int l = 0; l < 4; ++l) {
        const Layer& ly = L[l];
        const int hd = ly.hd, cin = ly.cin, nch = ly.nch, Cpad = nch * 32;
        const int Mt2G = hd / 16;    // gates: Mtiles = 2hd/16, pairs = hd/16
        const int Mt2C = hd / 32;    // cand : Mtiles =  hd/16, pairs = hd/32

        if (l == 0 || l == 1) {
            int nEl = Nn * hd;
            k_fill0<<<(nEl + 255) / 256, 256, 0, stream>>>(hstate, nEl);
        } else if (l == 2) {   // decoder layer0 init = encoder layer1 final h
            hipMemcpyAsync(hstate, h1save, (size_t)Nn * 64 * 4, hipMemcpyDeviceToDevice, stream);
        } else {               // decoder layer1 init = encoder layer0 final h
            hipMemcpyAsync(hstate, h0save, (size_t)Nn * 32 * 4, hipMemcpyDeviceToDevice, stream);
        }

        // init cat/cat2: zero pad + guard row, h-region from initial state
        int initEl = (Nn + 1) * Cpad;
        k_init_cat<<<(initEl + 255) / 256, 256, 0, stream>>>(hstate, cat, cat2, cin, hd, Cpad);

        const int axBlocks = (Nn * cin + 255) / 256;
        const int gBlocks  = Mt2G * NT / 8;
        const int cBlocks  = Mt2C * NT / 8;

        for (int t = 0; t < Tt; ++t) {
            const _Float16* sin_t = ly.sin + (size_t)t * Nn * cin;
            _Float16* sout_t = ly.sout + (size_t)t * Nn * hd;
            k_assemble_x<<<axBlocks, 256, 0, stream>>>(sin_t, cat, cat2, cin, Cpad);
            switch (nch) {
            case 2:
                k_gates<2><<<gBlocks, 256, 0, stream>>>(cat, packs[2 * l], ly.bg, hstate, cat2, ubuf, Mt2G, cin, hd);
                k_cand<2><<<cBlocks, 256, 0, stream>>>(cat2, packs[2 * l + 1], ly.bc, ubuf, hstate, sout_t, cat, Mt2C, cin, hd);
                break;
            case 3:
                k_gates<3><<<gBlocks, 256, 0, stream>>>(cat, packs[2 * l], ly.bg, hstate, cat2, ubuf, Mt2G, cin, hd);
                k_cand<3><<<cBlocks, 256, 0, stream>>>(cat2, packs[2 * l + 1], ly.bc, ubuf, hstate, sout_t, cat, Mt2C, cin, hd);
                break;
            default:
                k_gates<4><<<gBlocks, 256, 0, stream>>>(cat, packs[2 * l], ly.bg, hstate, cat2, ubuf, Mt2G, cin, hd);
                k_cand<4><<<cBlocks, 256, 0, stream>>>(cat2, packs[2 * l + 1], ly.bc, ubuf, hstate, sout_t, cat, Mt2C, cin, hd);
                break;
            }
        }
        if (l == 0) hipMemcpyAsync(h0save, hstate, (size_t)Nn * 32 * 4, hipMemcpyDeviceToDevice, stream);
        if (l == 1) hipMemcpyAsync(h1save, hstate, (size_t)Nn * 64 * 4, hipMemcpyDeviceToDevice, stream);
    }

    k_final<<<(Tt * Nn + 255) / 256, 256, 0, stream>>>(seq3, fw, fb, out);
}
